// Int8Linear_28123445854778
// MI455X (gfx1250) — compile-verified, matrix operand order checked by reading, not run on hardware
//
#include <hip/hip_runtime.h>

// ---------------------------------------------------------------------------
// Int8-dequant linear: out[t,o] = ( sum_i x[t,i] * w_int8[o,i] ) * scale[o]
// X: [8192, 4096] bf16   W: [11008, 4096] int32 (int8 range)   S: [11008] f32
// O: [8192, 11008] bf16
//
// CDNA5: v_wmma_f32_16x16x32_bf16, double-buffered LDS, async global->LDS
// (ASYNCcnt) for the A tile, register-pipelined int8->bf16 dequant for W,
// f32 accumulate, per-channel scale epilogue.
// ---------------------------------------------------------------------------

typedef __bf16 bf16_t;
typedef __attribute__((ext_vector_type(16))) __bf16 v16bf;
typedef __attribute__((ext_vector_type(8)))  __bf16 v8bf;
typedef __attribute__((ext_vector_type(4)))  __bf16 v4bf;
typedef __attribute__((ext_vector_type(8)))  float  v8f;
typedef __attribute__((ext_vector_type(4)))  int    v4i;

#define TOKENS_ 8192
#define IN_     4096
#define OUT_    11008

#define BM 256      // M tile per workgroup (4 waves x 64)
#define BN 128      // N tile per workgroup (2 waves x 64)
#define BK 64       // K per LDS phase (two WMMA k-steps)
#define LDT 72      // LDS row stride in bf16 elems (64 + 8 pad = 144 B, 16B-aligned)

#define A_ELEMS (BM * LDT)          // 18432 bf16 per A buffer
#define W_ELEMS (BN * LDT)          //  9216 bf16 per W buffer
#define SMEM_BYTES ((2 * A_ELEMS + 2 * W_ELEMS) * 2)   // 110592 B

union FragU {
    v16bf v;
    struct { v8bf lo, hi; } h;
};

extern __shared__ __attribute__((aligned(32))) char smem_raw[];

__device__ __forceinline__ void async_copy_A(const bf16_t* __restrict__ X,
                                             int tileM, int kp, int buf,
                                             int ar, int ac)
{
    // 256 rows x 64 bf16 = 32 KB; 8 x b128 per thread, ASYNCcnt-tracked.
#pragma unroll
    for (int p = 0; p < 8; ++p) {
        const int row = p * 32 + ar;
        const unsigned long long ga =
            (unsigned long long)(const void*)(X + (size_t)(tileM + row) * IN_ + kp + ac * 8);
        const unsigned lofs = (unsigned)(buf * A_ELEMS * 2 + (row * LDT + ac * 8) * 2);
        asm volatile("global_load_async_to_lds_b128 %0, %1, off"
                     :
                     : "v"(lofs), "v"(ga)
                     : "memory");
    }
}

__global__ __launch_bounds__(256)
void int8_linear_wmma_kernel(const bf16_t* __restrict__ X,
                             const int*    __restrict__ W,
                             const float*  __restrict__ S,
                             bf16_t*       __restrict__ O)
{
    bf16_t* lds_a = (bf16_t*)smem_raw;                 // [2][A_ELEMS]
    bf16_t* lds_w = (bf16_t*)smem_raw + 2 * A_ELEMS;   // [2][W_ELEMS]

    const int tid   = threadIdx.x;
    const int lane  = tid & 31;
    const int wave  = tid >> 5;     // 0..7
    const int wm    = wave & 3;     // 4 waves along M (64 rows each)
    const int wn    = wave >> 2;    // 2 waves along N (64 cols each)
    const int lrow  = lane & 15;
    const int lhalf = lane >> 4;

    const int tileN = blockIdx.x * BN;   // 86 tiles
    const int tileM = blockIdx.y * BM;   // 32 tiles

    // global-load thread mapping
    const int ar = tid >> 3;   // 0..31 : A row within 32-row pass
    const int ac = tid & 7;    // 0..7  : 16B column block in A
    const int wr = tid >> 4;   // 0..15 : W row within 16-row pass
    const int wc = tid & 15;   // 0..15 : 16B (4-int) column block in W

    v8f acc[4][4];
#pragma unroll
    for (int i = 0; i < 4; ++i)
#pragma unroll
        for (int j = 0; j < 4; ++j)
            acc[i][j] = (v8f){0.f, 0.f, 0.f, 0.f, 0.f, 0.f, 0.f, 0.f};

    v4i wreg[8];

    // ---------------- Prologue: fill buffer 0 for kp = 0 ----------------
    async_copy_A(X, tileM, 0, 0, ar, ac);
#pragma unroll
    for (int p = 0; p < 8; ++p) {
        const int row = p * 16 + wr;
        wreg[p] = *(const v4i*)(W + (size_t)(tileN + row) * IN_ + wc * 4);
    }
#pragma unroll
    for (int p = 0; p < 8; ++p) {
        const int row = p * 16 + wr;
        v4bf pk;
        pk[0] = (bf16_t)(float)wreg[p][0];   // exact: |w| <= 128
        pk[1] = (bf16_t)(float)wreg[p][1];
        pk[2] = (bf16_t)(float)wreg[p][2];
        pk[3] = (bf16_t)(float)wreg[p][3];
        *(v4bf*)(&lds_w[row * LDT + wc * 4]) = pk;
    }
    asm volatile("s_wait_asynccnt 0x0" ::: "memory");
    __syncthreads();

    // ---------------- Main K loop: 64 double-buffered phases ----------------
    for (int kp = 0; kp < IN_; kp += BK) {
        const int cur  = (kp >> 6) & 1;
        const int nxt  = cur ^ 1;
        const bool more = (kp + BK) < IN_;

        // Issue next phase's global traffic before computing (overlaps WMMA).
        if (more) {
            async_copy_A(X, tileM, kp + BK, nxt, ar, ac);
#pragma unroll
            for (int p = 0; p < 8; ++p) {
                const int row = p * 16 + wr;
                wreg[p] = *(const v4i*)(W + (size_t)(tileN + row) * IN_ + (kp + BK) + wc * 4);
            }
        }

        const bf16_t* abuf = lds_a + cur * A_ELEMS;
        const bf16_t* wbuf = lds_w + cur * W_ELEMS;

        // Two WMMA k-steps of 32 over the 64-wide LDS tiles.
#pragma unroll
        for (int ks = 0; ks < BK; ks += 32) {
            FragU afr[4];
#pragma unroll
            for (int i = 0; i < 4; ++i) {
                const int base = (wm * 64 + i * 16 + lrow) * LDT + ks + lhalf * 8;
                afr[i].h.lo = *(const v8bf*)(&abuf[base]);
                afr[i].h.hi = *(const v8bf*)(&abuf[base + 16]);
            }
            FragU bfr[4];
#pragma unroll
            for (int j = 0; j < 4; ++j) {
                const int base = (wn * 64 + j * 16 + lrow) * LDT + ks + lhalf * 8;
                bfr[j].h.lo = *(const v8bf*)(&wbuf[base]);
                bfr[j].h.hi = *(const v8bf*)(&wbuf[base + 16]);
            }
#pragma unroll
            for (int i = 0; i < 4; ++i)
#pragma unroll
                for (int j = 0; j < 4; ++j)
                    acc[i][j] = __builtin_amdgcn_wmma_f32_16x16x32_bf16(
                        false, afr[i].v, false, bfr[j].v,
                        (short)0, acc[i][j], false, false);
        }

        // Convert + park next W tile; wait for async A; handshake.
        if (more) {
            bf16_t* wdst = lds_w + nxt * W_ELEMS;
#pragma unroll
            for (int p = 0; p < 8; ++p) {
                const int row = p * 16 + wr;
                v4bf pk;
                pk[0] = (bf16_t)(float)wreg[p][0];
                pk[1] = (bf16_t)(float)wreg[p][1];
                pk[2] = (bf16_t)(float)wreg[p][2];
                pk[3] = (bf16_t)(float)wreg[p][3];
                *(v4bf*)(&wdst[row * LDT + wc * 4]) = pk;
            }
            asm volatile("s_wait_asynccnt 0x0" ::: "memory");
        }
        __syncthreads();
    }

    // ---------------- Epilogue: per-channel scale, f32 -> bf16 ----------------
    // C/D layout: VGPR r of lane l holds (M = r + 8*(l/16), N = l%16).
#pragma unroll
    for (int j = 0; j < 4; ++j) {
        const int n = tileN + wn * 64 + j * 16 + lrow;
        const float s = S[n];
#pragma unroll
        for (int i = 0; i < 4; ++i) {
            const int mbase = tileM + wm * 64 + i * 16 + lhalf * 8;
#pragma unroll
            for (int r = 0; r < 8; ++r) {
                O[(size_t)(mbase + r) * OUT_ + n] = (bf16_t)(acc[i][j][r] * s);
            }
        }
    }
}

extern "C" void kernel_launch(void* const* d_in, const int* in_sizes, int n_in,
                              void* d_out, int out_size, void* d_ws, size_t ws_size,
                              hipStream_t stream) {
    const bf16_t* X = (const bf16_t*)d_in[0];   // [8192, 4096] bf16
    const int*    W = (const int*)d_in[1];      // [11008, 4096] int32 (int8 range)
    const float*  S = (const float*)d_in[2];    // [11008] f32
    bf16_t*       O = (bf16_t*)d_out;           // [8192, 11008] bf16

    (void)hipFuncSetAttribute((const void*)int8_linear_wmma_kernel,
                              hipFuncAttributeMaxDynamicSharedMemorySize,
                              SMEM_BYTES);

    dim3 grid(OUT_ / BN, TOKENS_ / BM);         // (86, 32)
    dim3 block(256);
    int8_linear_wmma_kernel<<<grid, block, SMEM_BYTES, stream>>>(X, W, S, O);
}
